// VAEEncoder_26972394619782
// MI455X (gfx1250) — matrix operand order repaired and checked by reading, not compiled, not run on hardware
//
#include <hip/hip_runtime.h>

// =====================================================================
// VAE encoder forward for MI455X (gfx1250), wave32, bf16 WMMA everywhere.
// All GEMMs lower to v_wmma_f32_16x16x32_bf16 with f32 accumulation.
// Weights are pre-transposed to [K][N] bf16 once, so the GEMM hot loop
// is global_load_b128 / ds_store_b128 / ds_load_b128 (A) / per-element B gathers / v_wmma.
// NOTE: requires ws_size >= ~830 MB (arena layout documented below).
// =====================================================================

typedef unsigned short u16;
typedef unsigned int   u32;
typedef unsigned long long u64;
typedef __attribute__((ext_vector_type(16))) __bf16 v16bf;
typedef __attribute__((ext_vector_type(8)))  float  v8f;

struct alignas(16) U4 { u32 x, y, z, w; };
union F16x16 { U4 q[2]; v16bf v; };   // 32B: one WMMA 16-bit A/B fragment
union U4x8   { U4 q; u16 s[8]; };

static __device__ __forceinline__ u16 f2b(float f) {
  u32 u = __float_as_uint(f);
  u32 r = (u + 0x7FFFu + ((u >> 16) & 1u)) >> 16;   // RNE
  return (u16)r;
}
static __device__ __forceinline__ float b2f(u16 h) {
  return __uint_as_float(((u32)h) << 16);
}
static __device__ __forceinline__ float gelu_f(float v) {
  return 0.5f * v * (1.f + erff(v * 0.70710678118f));
}

// ---------------------------------------------------------------------
// f32 [N][K] -> bf16 [K][N] transposing conversion (one-time per weight)
// ---------------------------------------------------------------------
__global__ void convT_k(const float* __restrict__ w, u16* __restrict__ wt,
                        int N, int K) {
  long i = (long)blockIdx.x * 256 + threadIdx.x;
  if (i < (long)N * K) {
    int n = (int)(i / K), k = (int)(i % K);
    wt[(long)k * N + n] = f2b(w[i]);
  }
}

// ---------------------------------------------------------------------
// Precompute: te_proj[32][256]  (tile LUT through LN+GELU+fusion W)
//             pos_bias[169][256] (col/row embeddings folded + efus_b)
// ---------------------------------------------------------------------
__global__ __launch_bounds__(256) void precompute_k(
    const float* __restrict__ emb_tile, const float* __restrict__ ltg, const float* __restrict__ ltb,
    const float* __restrict__ emb_col,  const float* __restrict__ lcg, const float* __restrict__ lcb,
    const float* __restrict__ emb_row,  const float* __restrict__ lrg, const float* __restrict__ lrb,
    const float* __restrict__ efW, const float* __restrict__ efb,
    float* __restrict__ te_proj, float* __restrict__ pos_bias)
{
  __shared__ float buf[58][128];
  __shared__ float red[128];
  const int tid = threadIdx.x;
  for (int r = 0; r < 58; ++r) {
    const float *src, *g, *bb;
    if (r < 32)      { src = emb_tile + r * 128;        g = ltg; bb = ltb; }
    else if (r < 45) { src = emb_col + (r - 32) * 128;  g = lcg; bb = lcb; }
    else             { src = emb_row + (r - 45) * 128;  g = lrg; bb = lrb; }
    float v = (tid < 128) ? src[tid] : 0.f;
    if (tid < 128) red[tid] = v;
    __syncthreads();
    for (int st = 64; st > 0; st >>= 1) { if (tid < st) red[tid] += red[tid + st]; __syncthreads(); }
    float mean = red[0] * (1.f / 128.f);
    __syncthreads();
    float d = v - mean;
    if (tid < 128) red[tid] = d * d;
    __syncthreads();
    for (int st = 64; st > 0; st >>= 1) { if (tid < st) red[tid] += red[tid + st]; __syncthreads(); }
    float var = red[0] * (1.f / 128.f);
    __syncthreads();
    if (tid < 128) {
      float yv = d * rsqrtf(var + 1e-5f) * g[tid] + bb[tid];
      buf[r][tid] = gelu_f(yv);
    }
    __syncthreads();
  }
  const int d = tid;  // output channel 0..255
  for (int v = 0; v < 32; ++v) {
    float s = 0.f;
    for (int j = 0; j < 128; ++j) s += buf[v][j] * efW[d * 384 + j];
    te_proj[v * 256 + d] = s;
  }
  for (int sp = 0; sp < 169; ++sp) {
    int c = sp % 13, r = sp / 13;
    float acc = efb[d];
    for (int j = 0; j < 128; ++j) acc += buf[32 + c][j] * efW[d * 384 + 128 + j];
    for (int j = 0; j < 128; ++j) acc += buf[45 + r][j] * efW[d * 384 + 256 + j];
    pos_bias[sp * 256 + d] = acc;
  }
}

// ---------------------------------------------------------------------
// embed_bf16[b*169+s][d] = te_proj[x[b,s]][d] + pos_bias[s][d]
// ---------------------------------------------------------------------
__global__ __launch_bounds__(256) void embed_k(const int* __restrict__ x,
                                               const float* __restrict__ te,
                                               const float* __restrict__ pb,
                                               u16* __restrict__ out)
{
  long i = (long)blockIdx.x * 256 + threadIdx.x;
  long bs = i >> 8; int d = (int)(i & 255);
  int s = (int)(bs % 169);
  int t = x[bs];
  out[i] = f2b(te[t * 256 + d] + pb[s * 256 + d]);
}

// ---------------------------------------------------------------------
// Generic bf16 GEMM:  C[M,N] = act(A[M,K]_bf16 @ Wt + bias[N])
//   A  : [M][K] row-major (activations)
//   Wt : [K][N] row-major (pre-transposed weight, row stride ldb == N)
// Requires M % 64 == 0, K % 64 == 0, N % 16 == 0 (true at every call site).
// 64x64 block tile, 64-K stage, 8 waves; each wave owns a 32x16 C strip
// (2 accumulators, shared B fragment). A loads are b128; B fragments are gathered per element.
// ---------------------------------------------------------------------
__global__ __launch_bounds__(256) void gemm_bf16_k(
    const u16* __restrict__ A, long lda,
    const u16* __restrict__ Bt, long ldb,
    const float* __restrict__ bias,
    float* __restrict__ Cf, u16* __restrict__ Cb, long ldc,
    int M, int N, int K, int act)
{
  __shared__ alignas(16) u16 sA[64][72];   // [m][k], 8-u16 pad -> 144B rows
  __shared__ alignas(16) u16 sB[64][72];   // [k][n], 8-u16 pad
  const int tid  = threadIdx.x;
  const int m0   = blockIdx.y * 64;
  const int n0   = blockIdx.x * 64;
  const int wave = tid >> 5;
  const int lane = tid & 31;
  const int wmB  = (wave & 1) * 32;        // wave row block (0/32)
  const int wn   = (wave >> 1) * 16;       // wave col block (0/16/32/48)
  const int fr   = tid >> 2;               // fill row 0..63 (A: m, B: k)
  const int fc   = (tid & 3) * 16;         // fill col group (A: k, B: n)

  const bool nok  = (n0 + fc) < N;                 // N % 16 == 0
  const int  bcol = nok ? (n0 + fc) : 0;
  const u16* Arow = A + (long)(m0 + fr) * lda + fc;
  const u16* Bp   = Bt + (long)fr * ldb + bcol;    // advances by 64*ldb per stage

  v8f acc0 = {0.f,0.f,0.f,0.f,0.f,0.f,0.f,0.f};
  v8f acc1 = {0.f,0.f,0.f,0.f,0.f,0.f,0.f,0.f};

  const int am = lane & 15;
  const int kb = (lane >> 4) * 8;          // per-lane K base within 32-chunk

  for (int k0 = 0; k0 < K; k0 += 64) {
    U4 a0 = *(const U4*)(Arow + k0);
    U4 a1 = *(const U4*)(Arow + k0 + 8);
    U4 b0 = *(const U4*)(Bp + (long)k0 * ldb);
    U4 b1 = *(const U4*)(Bp + (long)k0 * ldb + 8);
    if (!nok) { b0.x = b0.y = b0.z = b0.w = 0u; b1 = b0; }
    if (k0 + 64 < K) {
      __builtin_prefetch(Arow + k0 + 64, 0, 3);
      __builtin_prefetch(Bp + (long)(k0 + 64) * ldb, 0, 3);
    }
    *(U4*)(&sA[fr][fc])     = a0;
    *(U4*)(&sA[fr][fc + 8]) = a1;
    *(U4*)(&sB[fr][fc])     = b0;
    *(U4*)(&sB[fr][fc + 8]) = b1;
    __syncthreads();
#pragma unroll
    for (int kc = 0; kc < 64; kc += 32) {
      F16x16 ub, ua0, ua1;
      { U4x8 t0, t1;                                     // lane = n, elements = k (table order)
        for (int j = 0; j < 8; ++j) { t0.s[j] = sB[kc + kb + j][wn + am]; t1.s[j] = sB[kc + 16 + kb + j][wn + am]; }
        ub.q[0] = t0.q; ub.q[1] = t1.q; }
      ua0.q[0] = *(const U4*)(&sA[wmB + am][kc + kb]);
      ua0.q[1] = *(const U4*)(&sA[wmB + am][kc + 16 + kb]);
      ua1.q[0] = *(const U4*)(&sA[wmB + 16 + am][kc + kb]);
      ua1.q[1] = *(const U4*)(&sA[wmB + 16 + am][kc + 16 + kb]);
      acc0 = __builtin_amdgcn_wmma_f32_16x16x32_bf16(false, ua0.v, false, ub.v,
                                                     (short)0, acc0, false, false);
      acc1 = __builtin_amdgcn_wmma_f32_16x16x32_bf16(false, ua1.v, false, ub.v,
                                                     (short)0, acc1, false, false);
    }
    __syncthreads();
  }
  // epilogue: lane -> (n = lane%16, m = r + 8*(lane/16))
  const int cn = n0 + wn + (lane & 15);
  if (cn < N) {
    float bv = bias ? bias[cn] : 0.f;
    const int mb = m0 + wmB + 8 * (lane >> 4);
#pragma unroll
    for (int sub = 0; sub < 2; ++sub) {
#pragma unroll
      for (int r = 0; r < 8; ++r) {
        float v = (sub ? acc1[r] : acc0[r]) + bv;
        if (act == 1)      v = fmaxf(v, 0.f);
        else if (act == 2) v = gelu_f(v);
        long off = (long)(mb + 16 * sub + r) * ldc + cn;
        if (Cf) Cf[off] = v;
        if (Cb) Cb[off] = f2b(v);
      }
    }
  }
}

// ---------------------------------------------------------------------
// GRU gate pointwise (torch GRUCell order r,z,n). 1024*512 elements.
// ---------------------------------------------------------------------
__global__ __launch_bounds__(256) void gru_gate_k(
    const float* __restrict__ gA,  // h @ Wh^T + bh   [1024,1536]
    const float* __restrict__ gB,  // e_s @ Wi^T + bi [1024,1536]
    float* __restrict__ hF, u16* __restrict__ hB, u16* __restrict__ hsB, int s)
{
  int idx = blockIdx.x * 256 + threadIdx.x;
  int b = idx >> 9, j = idx & 511;
  const float* a = gA + (long)b * 1536;
  const float* g = gB + (long)b * 1536;
  float r = 1.f / (1.f + __expf(-(g[j]         + a[j])));
  float z = 1.f / (1.f + __expf(-(g[512 + j]   + a[512 + j])));
  float n = tanhf(g[1024 + j] + r * a[1024 + j]);
  float h  = hF[idx];
  float hn = (1.f - z) * n + z * h;
  hF[idx] = hn;
  u16 hb = f2b(hn);
  hB[idx] = hb;
  hsB[((long)b * 169 + s) * 512 + j] = hb;
}

__global__ void zeroh_k(float* hF, u16* hB) {
  int i = blockIdx.x * 256 + threadIdx.x;
  hF[i] = 0.f; hB[i] = 0;
}

// ---------------------------------------------------------------------
// Row-wise LayerNorm (+optional residual, +optional GELU).
// blockDim.x == N (256 or 512). Inputs f32 or bf16; outputs f32 and/or bf16.
// ---------------------------------------------------------------------
__global__ __launch_bounds__(512) void ln_act_k(
    const float* __restrict__ Xf, const u16* __restrict__ Xb,
    const float* __restrict__ R,
    const float* __restrict__ g, const float* __restrict__ bta,
    int act, float* __restrict__ outF, u16* __restrict__ outB, int N)
{
  __shared__ float red[512];
  const long row = blockIdx.x;
  const int  t   = threadIdx.x;
  const long off = row * N + t;
  float x = Xf ? Xf[off] : b2f(Xb[off]);
  if (R) x += R[off];
  red[t] = x; __syncthreads();
  for (int s = N >> 1; s > 0; s >>= 1) { if (t < s) red[t] += red[t + s]; __syncthreads(); }
  float mean = red[0] / (float)N;
  __syncthreads();
  float d = x - mean;
  red[t] = d * d; __syncthreads();
  for (int s = N >> 1; s > 0; s >>= 1) { if (t < s) red[t] += red[t + s]; __syncthreads(); }
  float var = red[0] / (float)N;
  __syncthreads();
  float y = d * rsqrtf(var + 1e-5f) * g[t] + bta[t];
  if (act == 2) y = gelu_f(y);
  if (outF) outF[off] = y;
  if (outB) outB[off] = f2b(y);
}

// ---------------------------------------------------------------------
// Attention: one workgroup per (batch, head). nh=2, hd=128, S=169 (pad 176/192).
// K stored feature-major (transposed) and V row-major in LDS; both WMMA
// stages load A fragments as b128 and gather B per element. Dynamic LDS ~119 KB.
// ---------------------------------------------------------------------
#define AT_KT   0                      // u16 [128][176]   45056 B
#define AT_V    45056                  // u16 [192][136]   52224 B
#define AT_SQ   (45056 + 52224)        // u16 [16][136]     4352 B
#define AT_SC   (AT_SQ + 4352)         // f32 [16][192]    12288 B
#define AT_PB   (AT_SC + 12288)        // u16 [16][208]     6656 B
#define AT_RED  (AT_PB + 6656)         // f32 [16][16]      1024 B
#define AT_RM   (AT_RED + 1024)        // f32 [16]            64 B
#define AT_RI   (AT_RM + 64)           // f32 [16]            64 B
#define AT_SMEM (AT_RI + 64)

__global__ __launch_bounds__(256) void attn_k(const u16* __restrict__ qkv,
                                              u16* __restrict__ o)
{
  extern __shared__ char smem[];
  u16*   Kt   = (u16*)(smem + AT_KT);
  u16*   V    = (u16*)(smem + AT_V);
  u16*   sq   = (u16*)(smem + AT_SQ);
  float* sc   = (float*)(smem + AT_SC);
  u16*   pb   = (u16*)(smem + AT_PB);
  float* red2 = (float*)(smem + AT_RED);
  float* rowm = (float*)(smem + AT_RM);
  float* rowi = (float*)(smem + AT_RI);

  const int b = blockIdx.x, h = blockIdx.y;
  const int tid = threadIdx.x, wave = tid >> 5, lane = tid & 31;
  const long base = (long)b * 169 * 768;

  // stage K (transposed: feature-major) and V (natural, padded to 192 rows)
  for (int idx = tid; idx < 176 * 16; idx += 256) {
    int s = idx >> 4, c8 = (idx & 15) * 8;
    U4 kq = {0u,0u,0u,0u}, vq = {0u,0u,0u,0u};
    if (s < 169) {
      kq = *(const U4*)(qkv + base + (long)s * 768 + 256 + h * 128 + c8);
      vq = *(const U4*)(qkv + base + (long)s * 768 + 512 + h * 128 + c8);
    }
    *(U4*)(&V[s * 136 + c8]) = vq;
    U4x8 t; t.q = kq;
#pragma unroll
    for (int j = 0; j < 8; ++j) Kt[(c8 + j) * 176 + s] = t.s[j];
  }
  for (int idx = tid; idx < 16 * 136; idx += 256) V[176 * 136 + idx] = 0;  // V pad rows
  __syncthreads();

  for (int qi = 0; qi < 11; ++qi) {
    for (int idx = tid; idx < 16 * 16; idx += 256) {   // stage q tile (b128)
      int r = idx >> 4, c8 = (idx & 15) * 8;
      int srow = qi * 16 + r;
      U4 q = {0u,0u,0u,0u};
      if (srow < 169) q = *(const U4*)(qkv + base + (long)srow * 768 + h * 128 + c8);
      *(U4*)(&sq[r * 136 + c8]) = q;
    }
    __syncthreads();

    for (int ct = wave; ct < 11; ct += 8) {            // scores = q @ K^T
      v8f acc = {0.f,0.f,0.f,0.f,0.f,0.f,0.f,0.f};
      const int am = lane & 15, kb = (lane >> 4) * 8;
#pragma unroll
      for (int kc = 0; kc < 4; ++kc) {
        const int k0 = kc * 32;
        F16x16 ua, ub;
        ua.q[0] = *(const U4*)(&sq[am * 136 + k0 + kb]);
        ua.q[1] = *(const U4*)(&sq[am * 136 + k0 + 16 + kb]);
        { U4x8 t0, t1;
          for (int j = 0; j < 8; ++j) { t0.s[j] = Kt[(k0 + kb + j) * 176 + ct * 16 + am]; t1.s[j] = Kt[(k0 + 16 + kb + j) * 176 + ct * 16 + am]; }
          ub.q[0] = t0.q; ub.q[1] = t1.q; }
        acc = __builtin_amdgcn_wmma_f32_16x16x32_bf16(false, ua.v, false, ub.v,
                                                      (short)0, acc, false, false);
      }
      const int n = lane & 15, mb = 8 * (lane >> 4);
#pragma unroll
      for (int r = 0; r < 8; ++r)
        sc[(mb + r) * 192 + ct * 16 + n] = acc[r] * 0.08838834764831845f; // 1/sqrt(128)
    }
    __syncthreads();

    { // softmax over 169 keys, 16 threads per row
      const int r = tid >> 4, c16 = tid & 15;
      float mx = -1e30f;
      for (int c = c16; c < 169; c += 16) mx = fmaxf(mx, sc[r * 192 + c]);
      red2[r * 16 + c16] = mx; __syncthreads();
      if (c16 == 0) { float m = red2[r * 16]; for (int j = 1; j < 16; ++j) m = fmaxf(m, red2[r * 16 + j]); rowm[r] = m; }
      __syncthreads();
      float m = rowm[r], sum = 0.f;
      for (int c = c16; c < 169; c += 16) sum += __expf(sc[r * 192 + c] - m);
      red2[r * 16 + c16] = sum; __syncthreads();
      if (c16 == 0) { float s2 = 0.f; for (int j = 0; j < 16; ++j) s2 += red2[r * 16 + j]; rowi[r] = 1.f / s2; }
      __syncthreads();
      float inv = rowi[r];
      bool rv = (qi * 16 + r) < 169;
      for (int c = c16; c < 192; c += 16) {
        float p = (rv && c < 169) ? __expf(sc[r * 192 + c] - m) * inv : 0.f;
        pb[r * 208 + c] = f2b(p);
      }
    }
    __syncthreads();

    { // o = probs @ V (each wave one 16-wide feature tile)
      const int n0 = wave * 16;
      const int am = lane & 15, kb = (lane >> 4) * 8;
      v8f acc = {0.f,0.f,0.f,0.f,0.f,0.f,0.f,0.f};
#pragma unroll
      for (int kc = 0; kc < 6; ++kc) {
        const int k0 = kc * 32;
        F16x16 ua, ub;
        ua.q[0] = *(const U4*)(&pb[am * 208 + k0 + kb]);
        ua.q[1] = *(const U4*)(&pb[am * 208 + k0 + 16 + kb]);
        { U4x8 t0, t1;
          for (int j = 0; j < 8; ++j) { t0.s[j] = V[(k0 + kb + j) * 136 + n0 + am]; t1.s[j] = V[(k0 + 16 + kb + j) * 136 + n0 + am]; }
          ub.q[0] = t0.q; ub.q[1] = t1.q; }
        acc = __builtin_amdgcn_wmma_f32_16x16x32_bf16(false, ua.v, false, ub.v,
                                                      (short)0, acc, false, false);
      }
      const int n = lane & 15, mb = 8 * (lane >> 4);
#pragma unroll
      for (int r = 0; r < 8; ++r) {
        int srow = qi * 16 + mb + r;
        if (srow < 169)
          o[((long)b * 169 + srow) * 256 + h * 128 + n0 + n] = f2b(acc[r]);
      }
    }
    __syncthreads();
  }
}

// ---------------------------------------------------------------------
// mean/max pool over S=169, emit bf16 [B, 512] = [mean | max]
// ---------------------------------------------------------------------
__global__ __launch_bounds__(256) void pool_k(const float* __restrict__ y, u16* __restrict__ hb) {
  const int b = blockIdx.x, d = threadIdx.x;
  const float* p = y + (long)b * 169 * 256 + d;
  float sum = 0.f, mx = -3.0e38f;
  for (int s = 0; s < 169; ++s) { float v = p[(long)s * 256]; sum += v; mx = fmaxf(mx, v); }
  hb[(long)b * 512 + d]       = f2b(sum * (1.f / 169.f));
  hb[(long)b * 512 + 256 + d] = f2b(mx);
}

// =====================================================================
// Host orchestration
// =====================================================================
extern "C" void kernel_launch(void* const* d_in, const int* in_sizes, int n_in,
                              void* d_out, int out_size, void* d_ws, size_t ws_size,
                              hipStream_t stream) {
  (void)in_sizes; (void)n_in; (void)out_size; (void)ws_size;
  const long BS = 1024L * 169;  // 173056

  // ---- inputs (setup_inputs order) ----
  const int*   x        = (const int*)  d_in[0];
  const float* emb_tile = (const float*)d_in[1];
  const float* ln_t_g   = (const float*)d_in[2];
  const float* ln_t_b   = (const float*)d_in[3];
  const float* emb_col  = (const float*)d_in[4];
  const float* ln_c_g   = (const float*)d_in[5];
  const float* ln_c_b   = (const float*)d_in[6];
  const float* emb_row  = (const float*)d_in[7];
  const float* ln_r_g   = (const float*)d_in[8];
  const float* ln_r_b   = (const float*)d_in[9];
  const float* efus_W   = (const float*)d_in[10];
  const float* efus_b   = (const float*)d_in[11];
  const float* gru_Wi   = (const float*)d_in[12];
  const float* gru_Wh   = (const float*)d_in[13];
  const float* gru_bi   = (const float*)d_in[14];
  const float* gru_bh   = (const float*)d_in[15];
  const float* rfc1_W   = (const float*)d_in[16];
  const float* rfc1_b   = (const float*)d_in[17];
  const float* rln_g    = (const float*)d_in[18];
  const float* rln_b    = (const float*)d_in[19];
  const float* rfc2_W   = (const float*)d_in[20];
  const float* rfc2_b   = (const float*)d_in[21];
  const float* t_inW    = (const float*)d_in[22];
  const float* t_inb    = (const float*)d_in[23];
  const float* t_outW   = (const float*)d_in[24];
  const float* t_outb   = (const float*)d_in[25];
  const float* t_ln1g   = (const float*)d_in[26];
  const float* t_ln1b   = (const float*)d_in[27];
  const float* t_ff1W   = (const float*)d_in[28];
  const float* t_ff1b   = (const float*)d_in[29];
  const float* t_ff2W   = (const float*)d_in[30];
  const float* t_ff2b   = (const float*)d_in[31];
  const float* t_ln2g   = (const float*)d_in[32];
  const float* t_ln2b   = (const float*)d_in[33];
  const float* fnorm_g  = (const float*)d_in[34];
  const float* fnorm_b  = (const float*)d_in[35];
  const float* ffc_W    = (const float*)d_in[36];
  const float* ffc_b    = (const float*)d_in[37];
  const float* ffc_lng  = (const float*)d_in[38];
  const float* ffc_lnb  = (const float*)d_in[39];
  const float* mu_W     = (const float*)d_in[40];
  const float* mu_b     = (const float*)d_in[41];
  const float* lv_W     = (const float*)d_in[42];
  const float* lv_b     = (const float*)d_in[43];

  // ---- workspace arena (~823 MB; deterministic layout, explicit reuse) ----
  char* base = (char*)d_ws;
  size_t off = 0;
  auto alloc = [&](size_t bytes) { size_t o = off; off += (bytes + 255) & ~size_t(255); return o; };
  const size_t oWI   = alloc(1536 * 256 * 2);   // Wi^T  [256][1536]
  const size_t oWH   = alloc(1536 * 512 * 2);   // Wh^T  [512][1536]
  const size_t oR1W  = alloc(512 * 512 * 2);    // [512][512]
  const size_t oR2W  = alloc(256 * 512 * 2);    // [512][256]
  const size_t oTIN  = alloc(2 * 768 * 256 * 2);// per layer [256][768]
  const size_t oTOUT = alloc(2 * 256 * 256 * 2);
  const size_t oF1W  = alloc(2 * 256 * 256 * 2);
  const size_t oF2W  = alloc(2 * 256 * 256 * 2);
  const size_t oFFCW = alloc(512 * 512 * 2);
  const size_t oMUW  = alloc(32 * 512 * 2);     // [512][32]
  const size_t oLVW  = alloc(32 * 512 * 2);
  const size_t oTE   = alloc(32 * 256 * 4);
  const size_t oPB   = alloc(169 * 256 * 4);
  const size_t oHF   = alloc(1024 * 512 * 4);
  const size_t oHB   = alloc(1024 * 512 * 2);
  const size_t oGA   = alloc(1024 * 1536 * 4);
  const size_t oGB   = alloc(1024 * 1536 * 4);
  const size_t oPOOL = alloc(1024 * 512 * 2);
  const size_t oH2   = alloc(1024 * 512 * 2);
  // big, reused regions
  const size_t oEMB  = alloc((size_t)BS * 256 * 2);   // embed_bf; overlaid by qkv later
  const size_t oHS   = alloc((size_t)BS * 512 * 2);   // hs_bf;    overlaid by qkv later
  const size_t oQKV  = oEMB;                          // BS*768*2 == EMB+HS exactly
  const size_t oR1   = alloc((size_t)BS * 512 * 2);   // r1_bf; reused as tmp256 f32 (same bytes)
  const size_t oY    = alloc((size_t)BS * 256 * 4);
  const size_t oYB   = alloc((size_t)BS * 256 * 2);
  const size_t oOB   = alloc((size_t)BS * 256 * 2);   // attn out bf16; reused as ff1_bf

  u16*  WI    = (u16*)(base + oWI);
  u16*  WH    = (u16*)(base + oWH);
  u16*  R1W   = (u16*)(base + oR1W);
  u16*  R2W   = (u16*)(base + oR2W);
  u16*  TINW  = (u16*)(base + oTIN);
  u16*  TOUTW = (u16*)(base + oTOUT);
  u16*  F1Wb  = (u16*)(base + oF1W);
  u16*  F2Wb  = (u16*)(base + oF2W);
  u16*  FFCWb = (u16*)(base + oFFCW);
  u16*  MUWb  = (u16*)(base + oMUW);
  u16*  LVWb  = (u16*)(base + oLVW);
  float* TE   = (float*)(base + oTE);
  float* PBp  = (float*)(base + oPB);
  float* hF   = (float*)(base + oHF);
  u16*  hB    = (u16*)(base + oHB);
  float* gA   = (float*)(base + oGA);
  float* gB   = (float*)(base + oGB);
  u16*  POOLb = (u16*)(base + oPOOL);
  u16*  H2b   = (u16*)(base + oH2);
  u16*  EMB   = (u16*)(base + oEMB);
  u16*  HS    = (u16*)(base + oHS);
  u16*  QKV   = (u16*)(base + oQKV);
  u16*  R1b   = (u16*)(base + oR1);
  float* T256 = (float*)(base + oR1);   // reuse (same byte count)
  float* Y    = (float*)(base + oY);
  u16*  YB    = (u16*)(base + oYB);
  u16*  OB    = (u16*)(base + oOB);
  u16*  FF1b  = (u16*)(base + oOB);     // reuse

  auto cvtT = [&](const float* src, u16* dst, int N, int K) {
    long n = (long)N * K;
    convT_k<<<dim3((unsigned)((n + 255) / 256)), 256, 0, stream>>>(src, dst, N, K);
  };
  auto gemm = [&](const u16* A, long lda, const u16* Bt, const float* bias,
                  float* Cf, u16* Cb, long ldc, int M, int N, int K, int act) {
    dim3 g((unsigned)((N + 63) / 64), (unsigned)(M / 64));
    gemm_bf16_k<<<g, 256, 0, stream>>>(A, lda, Bt, (long)N, bias, Cf, Cb, ldc, M, N, K, act);
  };

  // ---- 0) weight conversion: f32 [N][K] -> bf16 [K][N] (one-time) ----
  cvtT(gru_Wi, WI,   1536, 256);
  cvtT(gru_Wh, WH,   1536, 512);
  cvtT(rfc1_W, R1W,  512, 512);
  cvtT(rfc2_W, R2W,  256, 512);
  for (int i = 0; i < 2; ++i) {
    cvtT(t_inW  + (size_t)i * 768 * 256, TINW  + (size_t)i * 768 * 256, 768, 256);
    cvtT(t_outW + (size_t)i * 256 * 256, TOUTW + (size_t)i * 256 * 256, 256, 256);
    cvtT(t_ff1W + (size_t)i * 256 * 256, F1Wb  + (size_t)i * 256 * 256, 256, 256);
    cvtT(t_ff2W + (size_t)i * 256 * 256, F2Wb  + (size_t)i * 256 * 256, 256, 256);
  }
  cvtT(ffc_W, FFCWb, 512, 512);
  cvtT(mu_W,  MUWb,  32, 512);
  cvtT(lv_W,  LVWb,  32, 512);

  // ---- 1) embedding folded into LUT + positional bias ----
  precompute_k<<<1, 256, 0, stream>>>(emb_tile, ln_t_g, ln_t_b, emb_col, ln_c_g, ln_c_b,
                                      emb_row, ln_r_g, ln_r_b, efus_W, efus_b, TE, PBp);
  embed_k<<<dim3((unsigned)(BS)), 256, 0, stream>>>(x, TE, PBp, EMB);

  // ---- 2) GRU scan: 169 sequential wide steps ----
  zeroh_k<<<2048, 256, 0, stream>>>(hF, hB);
  for (int s = 0; s < 169; ++s) {
    gemm(hB, 512, WH, gru_bh, gA, nullptr, 1536, 1024, 1536, 512, 0);
    gemm(EMB + (long)s * 256, 169L * 256, WI, gru_bi, gB, nullptr, 1536, 1024, 1536, 256, 0);
    gru_gate_k<<<2048, 256, 0, stream>>>(gA, gB, hF, hB, HS, s);
  }

  // ---- 3) recurrent head: gelu(ln(hs@rfc1+b)) @ rfc2 + b ----
  gemm(HS, 512, R1W, rfc1_b, nullptr, R1b, 512, (int)BS, 512, 512, 0);
  ln_act_k<<<dim3((unsigned)BS), 512, 0, stream>>>(nullptr, R1b, nullptr, rln_g, rln_b, 2,
                                                   nullptr, R1b, 512);
  gemm(R1b, 512, R2W, rfc2_b, Y, YB, 256, (int)BS, 256, 512, 0);

  // ---- 4) transformer: 2 post-norm layers, nh=2, ReLU FFN ----
  for (int i = 0; i < 2; ++i) {
    gemm(YB, 256, TINW + (size_t)i * 768 * 256, t_inb + i * 768,
         nullptr, QKV, 768, (int)BS, 768, 256, 0);
    attn_k<<<dim3(1024, 2), 256, AT_SMEM, stream>>>(QKV, OB);
    gemm(OB, 256, TOUTW + (size_t)i * 256 * 256, t_outb + i * 256,
         T256, nullptr, 256, (int)BS, 256, 256, 0);
    ln_act_k<<<dim3((unsigned)BS), 256, 0, stream>>>(T256, nullptr, Y, t_ln1g + i * 256,
                                                     t_ln1b + i * 256, 0, Y, YB, 256);
    gemm(YB, 256, F1Wb + (size_t)i * 256 * 256, t_ff1b + i * 256,
         nullptr, FF1b, 256, (int)BS, 256, 256, 1 /*relu*/);
    gemm(FF1b, 256, F2Wb + (size_t)i * 256 * 256, t_ff2b + i * 256,
         T256, nullptr, 256, (int)BS, 256, 256, 0);
    ln_act_k<<<dim3((unsigned)BS), 256, 0, stream>>>(T256, nullptr, Y, t_ln2g + i * 256,
                                                     t_ln2b + i * 256, 0, Y, YB, 256);
  }

  // ---- 5) final norm, pooling, head ----
  ln_act_k<<<dim3((unsigned)BS), 256, 0, stream>>>(Y, nullptr, nullptr, fnorm_g, fnorm_b, 0,
                                                   Y, nullptr, 256);
  pool_k<<<1024, 256, 0, stream>>>(Y, POOLb);
  gemm(POOLb, 512, FFCWb, ffc_b, nullptr, H2b, 512, 1024, 512, 512, 0);
  ln_act_k<<<1024, 512, 0, stream>>>(nullptr, H2b, nullptr, ffc_lng, ffc_lnb, 2,
                                     nullptr, H2b, 512);
  float* out_mu = (float*)d_out;
  float* out_lv = out_mu + 1024L * 32;
  gemm(H2b, 512, MUWb, mu_b, out_mu, nullptr, 32, 1024, 32, 512, 0);
  gemm(H2b, 512, LVWb, lv_b, out_lv, nullptr, 32, 1024, 32, 512, 0);
}
